// DeformableAttention_25563645346599
// MI455X (gfx1250) — matrix operand order repaired and checked by reading, not compiled, not run on hardware
//
#include <hip/hip_runtime.h>

// ---------------------------------------------------------------------------
// Deformable attention on gfx1250 (MI455X). Shapes fixed by setup_inputs():
//   B=4, Q=1024, D=256, H=W=128, NUM_HEADS=8, NUM_POINTS=4
// Pipeline: prep (offsets/attn/coords) -> WMMA value-proj GEMM ->
//           L2-resident bilinear gather -> WMMA output GEMM.
// ---------------------------------------------------------------------------

typedef __attribute__((ext_vector_type(16))) _Float16 v16h;
typedef __attribute__((ext_vector_type(8)))  _Float16 v8h;
typedef __attribute__((ext_vector_type(2)))  _Float16 v2h;
typedef __attribute__((ext_vector_type(8)))  float    v8f;

#define B_   4
#define Q_   1024
#define D_   256
#define NH   8
#define NP   4
#define H_   128
#define W_   128
#define HW_  (H_ * W_)
#define BQ_  (B_ * Q_)
#define NS_  (NH * NP)        // 32 samples per query

// ---------------------------------------------------------------------------
// Kernel A: per-query offsets + attention softmax + pixel coords.
// One block (128 threads) per (b,q). 96 dot products of length 256.
// ---------------------------------------------------------------------------
__global__ __launch_bounds__(128)
void prep_kernel(const float* __restrict__ query,
                 const float* __restrict__ refpts,
                 const float* __restrict__ W_off,  const float* __restrict__ b_off,
                 const float* __restrict__ W_attn, const float* __restrict__ b_attn,
                 float* __restrict__ sampxy,  // [BQ][NS][2] pixel coords
                 float* __restrict__ attnw)   // [BQ][NS] softmaxed weights
{
    const int bq = blockIdx.x;
    const int t  = threadIdx.x;

    __shared__ float qs[D_];
    __shared__ float offs[NS_ * 2];
    __shared__ float logits[NS_];

    for (int i = t; i < D_; i += 128) qs[i] = query[bq * D_ + i];
    __syncthreads();

    if (t < NS_ * 2) {                       // 64 offset outputs
        float acc = b_off[t];
        for (int k = 0; k < D_; ++k) acc += qs[k] * W_off[k * (NS_ * 2) + t];
        offs[t] = acc;
    } else if (t < NS_ * 2 + NS_) {          // 32 attention logits
        const int j = t - NS_ * 2;
        float acc = b_attn[j];
        for (int k = 0; k < D_; ++k) acc += qs[k] * W_attn[k * NS_ + j];
        logits[j] = acc;
    }
    __syncthreads();

    if (t < NS_) {
        const int j = t, h = j >> 2;
        const float l0 = logits[h * 4 + 0], l1 = logits[h * 4 + 1];
        const float l2 = logits[h * 4 + 2], l3 = logits[h * 4 + 3];
        const float m  = fmaxf(fmaxf(l0, l1), fmaxf(l2, l3));
        const float inv = 1.0f / (__expf(l0 - m) + __expf(l1 - m) +
                                  __expf(l2 - m) + __expf(l3 - m));
        const float w = __expf(logits[j] - m) * inv;

        const float rx = refpts[bq * 2 + 0];
        const float ry = refpts[bq * 2 + 1];
        // loc = clip(ref + off*0.1, 0, 1)*2 - 1 ; align_corners=False map
        const float gx = fminf(fmaxf(rx + offs[j * 2 + 0] * 0.1f, 0.f), 1.f) * 2.f - 1.f;
        const float gy = fminf(fmaxf(ry + offs[j * 2 + 1] * 0.1f, 0.f), 1.f) * 2.f - 1.f;
        const float px = ((gx + 1.f) * (float)W_ - 1.f) * 0.5f;
        const float py = ((gy + 1.f) * (float)H_ - 1.f) * 0.5f;

        const int s = bq * NS_ + j;
        sampxy[s * 2 + 0] = px;
        sampxy[s * 2 + 1] = py;
        attnw[s] = w;
    }
}

// ---------------------------------------------------------------------------
// WMMA GEMM: C[M,N] = A[M,K] * B[K,N] + bias[N]   (f32 accumulate)
// Workgroup tile 64x128, 8 waves as 2(M) x 4(N); each wave owns 32x32 via
// 2x2 register blocking: 2 A frags + 2 B frags (8 ds_load_b128) feed 4
// v_wmma_f32_16x16x32_f16 per K-step. Double-buffered LDS. Staging is
// phase-separated (all global loads clause first, then cvt, then LDS
// stores) so no per-element loadcnt waits; B pairs along K pack into one
// v_cvt_pk + ds_store_b32 each with a 2-way-conflict scatter pattern.
// ---------------------------------------------------------------------------
#define TM 64
#define TN 128
#define TK 32
#define LDA_S 40   // halfs: 80B row pitch, rows 16B-aligned
#define LDB_S 40   // halfs: BsT[n][k] pitch

__device__ inline v8h load8_as_half(const float* __restrict__ p) {
    const float4 u = ((const float4*)p)[0];
    const float4 w = ((const float4*)p)[1];
    v8h h;
    h[0] = (_Float16)u.x; h[1] = (_Float16)u.y;
    h[2] = (_Float16)u.z; h[3] = (_Float16)u.w;
    h[4] = (_Float16)w.x; h[5] = (_Float16)w.y;
    h[6] = (_Float16)w.z; h[7] = (_Float16)w.w;
    return h;
}
__device__ inline v8h load8_as_half(const _Float16* __restrict__ p) {
    return *(const v8h*)p;
}

#define NBPAIR ((TK * TN) / 512)   // 8 k-pairs per thread

template <typename TA>
__device__ inline void stage_tiles(const TA* __restrict__ A,
                                   const float* __restrict__ Bm,
                                   _Float16 (* __restrict__ As)[LDA_S],
                                   _Float16 (* __restrict__ BsT)[LDB_S],
                                   int t, int blockRow, int blockCol,
                                   int N, int K, int k0)
{
    // ---- phase 1: issue ALL global loads (one clause, one wait) ----
    const int rA  = t >> 2;
    const int cA8 = (t & 3) << 3;
    const v8h ha = load8_as_half(&A[(blockRow + rA) * K + k0 + cA8]);

    float blo[NBPAIR], bhi[NBPAIR];
    #pragma unroll
    for (int i = 0; i < NBPAIR; ++i) {
        const int p  = t + i * 256;
        const int r2 = (p >> 7) << 1;    // even k within tile
        const int c  = p & 127;          // n within tile
        const float* src = &Bm[(k0 + r2) * N + blockCol + c];
        blo[i] = src[0];
        bhi[i] = src[N];
    }
    // prefetch tile after next (global_prefetch_b8)
    if (k0 + TK < K)
        __builtin_prefetch(&A[(blockRow + rA) * K + k0 + TK + cA8], 0, 0);

    // ---- phase 2: convert + LDS stores ----
    *(v8h*)&As[rA][cA8] = ha;
    #pragma unroll
    for (int i = 0; i < NBPAIR; ++i) {
        const int p  = t + i * 256;
        const int r2 = (p >> 7) << 1;
        const int c  = p & 127;
        v2h pr;
        pr[0] = (_Float16)blo[i];
        pr[1] = (_Float16)bhi[i];
        *(v2h*)&BsT[c][r2] = pr;        // transposed: [n][k], k-pair packed
    }
}

__device__ inline v16h frag_from_row(const _Float16* __restrict__ p0,
                                     const _Float16* __restrict__ p1) {
    const v8h lo = *(const v8h*)p0;
    const v8h hi = *(const v8h*)p1;
    return __builtin_shufflevector(lo, hi,
        0, 1, 2, 3, 4, 5, 6, 7, 8, 9, 10, 11, 12, 13, 14, 15);
}

template <typename TA>
__global__ __launch_bounds__(256)
void wmma_gemm(const TA* __restrict__ A, const float* __restrict__ Bm,
               const float* __restrict__ bias, float* __restrict__ C,
               int M, int N, int K)
{
    __shared__ alignas(16) _Float16 AsBuf[2][TM][LDA_S];
    __shared__ alignas(16) _Float16 BsBuf[2][TN][LDB_S];

    const int t    = threadIdx.x;
    const int lane = t & 31;
    const int wave = t >> 5;
    const int row0 = (wave & 1) * 32;   // wave's M offset within tile
    const int col0 = (wave >> 1) * 32;  // wave's N offset within tile
    const int blockRow = blockIdx.x * TM;
    const int blockCol = blockIdx.y * TN;

    v8f c00 = {}; v8f c01 = {}; v8f c10 = {}; v8f c11 = {};

    const int n16 = lane & 15;          // fragment row/col within half-wave
    const int kbA = (lane >> 4) * 8;    // A layout: K base per lane half
    const int kbB = (lane >> 4) * 8;    // B layout: K base per lane half

    stage_tiles(A, Bm, AsBuf[0], BsBuf[0], t, blockRow, blockCol, N, K, 0);
    __syncthreads();

    const int nk = K / TK;
    for (int kk = 0; kk < nk; ++kk) {
        const int buf = kk & 1;
        if (kk + 1 < nk)
            stage_tiles(A, Bm, AsBuf[buf ^ 1], BsBuf[buf ^ 1],
                        t, blockRow, blockCol, N, K, (kk + 1) * TK);

        // --- fragments: pure ds_load_b128 traffic (8 loads, 4 WMMAs) ---
        const _Float16* a0r = &AsBuf[buf][row0 + n16][0];
        const _Float16* a1r = &AsBuf[buf][row0 + 16 + n16][0];
        const v16h a0 = frag_from_row(a0r + kbA, a0r + 16 + kbA);
        const v16h a1 = frag_from_row(a1r + kbA, a1r + 16 + kbA);

        const _Float16* b0p = &BsBuf[buf][col0 + n16][kbB];
        const _Float16* b1p = &BsBuf[buf][col0 + 16 + n16][kbB];
        const v16h b0 = frag_from_row(b0p, b0p + 16);
        const v16h b1 = frag_from_row(b1p, b1p + 16);

        c00 = __builtin_amdgcn_wmma_f32_16x16x32_f16(false, a0, false, b0,
                                                     (short)0, c00, false, false);
        c01 = __builtin_amdgcn_wmma_f32_16x16x32_f16(false, a0, false, b1,
                                                     (short)0, c01, false, false);
        c10 = __builtin_amdgcn_wmma_f32_16x16x32_f16(false, a1, false, b0,
                                                     (short)0, c10, false, false);
        c11 = __builtin_amdgcn_wmma_f32_16x16x32_f16(false, a1, false, b1,
                                                     (short)0, c11, false, false);
        __syncthreads();
    }

    // --- store D per 32-bit C/D layout: VGPR r -> M = r + 8*(lane>=16) ---
    const int mh = (lane >> 4) * 8;
    const int ca = blockCol + col0 + n16;
    const int cb = ca + 16;
    const float biasA = bias[ca];
    const float biasB = bias[cb];
    #pragma unroll
    for (int r = 0; r < 8; ++r) {
        const int rowTop = blockRow + row0 + mh + r;
        const int rowBot = rowTop + 16;
        C[rowTop * N + ca] = c00[r] + biasA;
        C[rowTop * N + cb] = c01[r] + biasB;
        C[rowBot * N + ca] = c10[r] + biasA;
        C[rowBot * N + cb] = c11[r] + biasB;
    }
}

// ---------------------------------------------------------------------------
// Kernel C: bilinear gather + point-weighted reduce. One block per (b,q);
// 8 heads x 32 lanes; lane owns channels 8c..8c+7 so every corner fetch is
// two coalesced global_load_b128 (img is L2-resident: 64MB < 192MB L2).
// Writes the output-GEMM A operand directly as one b128 f16 store.
// ---------------------------------------------------------------------------
__global__ __launch_bounds__(256)
void sample_kernel(const float* __restrict__ img,       // [B][HW][D] f32
                   const float* __restrict__ sampxy,    // [BQ][NS][2]
                   const float* __restrict__ attnw,     // [BQ][NS]
                   _Float16* __restrict__ weighted)     // [BQ][NH*D] f16
{
    const int bq = blockIdx.x;
    const int b  = bq >> 10;            // Q_ = 1024
    const int t  = threadIdx.x;
    const int h  = t >> 5;
    const int c8 = (t & 31) << 3;       // channel base

    float acc[8] = {0.f, 0.f, 0.f, 0.f, 0.f, 0.f, 0.f, 0.f};

    #pragma unroll
    for (int p = 0; p < NP; ++p) {
        const int s = bq * NS_ + h * NP + p;
        const float sx = sampxy[s * 2 + 0];
        const float sy = sampxy[s * 2 + 1];
        const float w  = attnw[s];

        const float x0f = floorf(sx), y0f = floorf(sy);
        const int   x0  = (int)x0f,   y0  = (int)y0f;
        const float wx1 = sx - x0f,   wy1 = sy - y0f;
        const float wxs[2] = {1.f - wx1, wx1};
        const float wys[2] = {1.f - wy1, wy1};

        #pragma unroll
        for (int cy = 0; cy < 2; ++cy) {
            #pragma unroll
            for (int cx = 0; cx < 2; ++cx) {
                const int xi = x0 + cx, yi = y0 + cy;
                const bool valid = (xi >= 0) & (xi < W_) & (yi >= 0) & (yi < H_);
                const float cw = w * wxs[cx] * wys[cy] * (valid ? 1.f : 0.f);
                const int xc = min(max(xi, 0), W_ - 1);
                const int yc = min(max(yi, 0), H_ - 1);
                // D_ = 256 -> shift, 32-bit addressing throughout
                const int base = ((b * HW_ + yc * W_ + xc) << 8) + c8;
                const float4 u = ((const float4*)(img + base))[0];
                const float4 v = ((const float4*)(img + base))[1];
                acc[0] += cw * u.x; acc[1] += cw * u.y;
                acc[2] += cw * u.z; acc[3] += cw * u.w;
                acc[4] += cw * v.x; acc[5] += cw * v.y;
                acc[6] += cw * v.z; acc[7] += cw * v.w;
            }
        }
    }

    v8h o;
    #pragma unroll
    for (int i = 0; i < 8; ++i) o[i] = (_Float16)acc[i];
    const int orow = bq * (NH * D_) + h * D_ + c8;
    *(v8h*)(weighted + orow) = o;
}

// ---------------------------------------------------------------------------
extern "C" void kernel_launch(void* const* d_in, const int* in_sizes, int n_in,
                              void* d_out, int out_size, void* d_ws, size_t ws_size,
                              hipStream_t stream)
{
    (void)in_sizes; (void)n_in; (void)out_size; (void)ws_size;

    const float* query = (const float*)d_in[0];
    const float* refp  = (const float*)d_in[1];
    const float* value = (const float*)d_in[2];
    const float* Woff  = (const float*)d_in[3];
    const float* boff  = (const float*)d_in[4];
    const float* Wattn = (const float*)d_in[5];
    const float* battn = (const float*)d_in[6];
    const float* Wv    = (const float*)d_in[7];
    const float* bv    = (const float*)d_in[8];
    const float* Wout  = (const float*)d_in[9];
    const float* bout  = (const float*)d_in[10];
    // d_in[11], d_in[12] = H, W (128; baked into constants above)

    char* ws = (char*)d_ws;
    const size_t imgBytes = (size_t)B_ * HW_ * D_ * sizeof(float);        // 64 MB
    const size_t wtdBytes = (size_t)BQ_ * NH * D_ * sizeof(_Float16);     // 16 MB
    float*    img      = (float*)ws;
    _Float16* weighted = (_Float16*)(ws + imgBytes);
    float*    sampxy   = (float*)(ws + imgBytes + wtdBytes);
    float*    attnw    = sampxy + (size_t)BQ_ * NS_ * 2;

    // 1) offsets / attention / sampling coords
    prep_kernel<<<BQ_, 128, 0, stream>>>(query, refp, Woff, boff, Wattn, battn,
                                         sampxy, attnw);

    // 2) value projection: img = value @ W_v + b_v   (M=65536, N=256, K=256)
    {
        dim3 grid((B_ * HW_) / TM, D_ / TN);
        wmma_gemm<float><<<grid, 256, 0, stream>>>(value, Wv, bv, img,
                                                   B_ * HW_, D_, D_);
    }

    // 3) bilinear gather + weighted reduce over points -> f16 A operand
    sample_kernel<<<BQ_, 256, 0, stream>>>(img, sampxy, attnw, weighted);

    // 4) output projection: out = weighted @ W_out + b_out (M=4096,N=256,K=2048)
    {
        dim3 grid(BQ_ / TM, D_ / TN);
        wmma_gemm<_Float16><<<grid, 256, 0, stream>>>(weighted, Wout, bout,
                                                      (float*)d_out,
                                                      BQ_, D_, NH * D_);
    }
}
